// EGNNModel_25305947308630
// MI455X (gfx1250) — compile-verified, matrix operand order checked by reading, not run on hardware
//
#include <hip/hip_runtime.h>
#include <hip/hip_bf16.h>

// ---------------- problem constants (from reference) ----------------
#define N_NODES 20000
#define N_EDGES 160000
#define DIM     128
#define NLAYERS 5
#define NGRAPH  64
#define LN_EPS  1e-5f

// LDS row strides (halves), padded so (stride/2) mod 64 banks has gcd 4 with 64
// -> no intra-16-lane bank conflicts on ds_load_b128 A-fragments; rows stay 16B aligned.
#define KP1 296   // edge GEMM1 input  (K = 288 used)
#define KP2 136   // K = 128 stages
#define KPN 264   // node GEMM1 input  (K = 256 used)

typedef __attribute__((ext_vector_type(16))) __bf16 v16bf;
typedef __attribute__((ext_vector_type(8)))  __bf16 v8bf;
typedef __attribute__((ext_vector_type(8)))  float  v8f;

// float -> bf16 round-to-nearest-even
__device__ __forceinline__ __bf16 f2bf(float f) {
    union { float f; unsigned u; } x; x.f = f;
    unsigned r = x.u + 0x7FFFu + ((x.u >> 16) & 1u);
    unsigned short hs = (unsigned short)(r >> 16);
    return __builtin_bit_cast(__bf16, hs);
}

// two floats -> packed bf16x2 (lo = a, hi = b), RNE
__device__ __forceinline__ unsigned pack2bf(float a, float b) {
    union { float f; unsigned u; } x, y; x.f = a; y.f = b;
    unsigned lo = (x.u + 0x7FFFu + ((x.u >> 16) & 1u)) >> 16;
    unsigned hi = (y.u + 0x7FFFu + ((y.u >> 16) & 1u)) & 0xFFFF0000u;
    return lo | hi;
}

// ---------------------------------------------------------------
// Wave-level GEMM: D[16 x 128] += A[16 x (32*kchunks)] * W[(32*kchunks) x 128]
// A: bf16 in LDS, row-major, row stride Kp halves (16B aligned rows).
// W: bf16, pre-packed in WMMA B-fragment order:
//    frag(kc,nt) = 512 halves; lane L holds halves [L*16 .. L*16+15].
// ---------------------------------------------------------------
__device__ __forceinline__ void wave_gemm(const __bf16* As, int Kp, int kchunks,
                                          const __bf16* Wp, int m0, int lane,
                                          v8f acc[8]) {
    const int lh  = lane >> 4;
    const int row = m0 + (lane & 15);   // A row owned by this lane (M = lane&15)
    for (int kc = 0; kc < kchunks; ++kc) {
        const __bf16* ap = As + row * Kp + kc * 32 + lh * 8;
        v8bf alo = *(const v8bf*)ap;          // K = base+0..7
        v8bf ahi = *(const v8bf*)(ap + 16);   // K = base+16..23
        v16bf a;
#pragma unroll
        for (int i = 0; i < 8; ++i) { a[i] = alo[i]; a[i + 8] = ahi[i]; }
        const __bf16* wbase = Wp + (size_t)kc * 8 * 512 + lane * 16;
#pragma unroll
        for (int nt = 0; nt < 8; ++nt) {
            v16bf b = *(const v16bf*)(wbase + nt * 512);
            acc[nt] = __builtin_amdgcn_wmma_f32_16x16x32_bf16(
                false, a, false, b, (short)0, acc[nt], false, false);
        }
    }
}

// bias add + LayerNorm(row over 128) + ReLU, in WMMA C/D register layout.
// Row M = v + 8*(lane>=16); col N = nt*16 + (lane&15).
__device__ __forceinline__ void bias_ln_relu(v8f acc[8], const float* bias,
                                             const float* gam, const float* bet,
                                             int lane) {
    const int col = lane & 15;
    float bb[8], gg[8], be[8];
#pragma unroll
    for (int nt = 0; nt < 8; ++nt) {
        int n = nt * 16 + col;
        bb[nt] = bias[n]; gg[nt] = gam[n]; be[nt] = bet[n];
    }
#pragma unroll
    for (int nt = 0; nt < 8; ++nt)
#pragma unroll
        for (int v = 0; v < 8; ++v) acc[nt][v] += bb[nt];

    const float inv = 1.0f / 128.0f;
#pragma unroll
    for (int v = 0; v < 8; ++v) {
        float s = 0.f, s2 = 0.f;
#pragma unroll
        for (int nt = 0; nt < 8; ++nt) { float x = acc[nt][v]; s += x; s2 += x * x; }
        for (int m = 1; m < 16; m <<= 1) {
            s  += __shfl_xor(s,  m, 32);
            s2 += __shfl_xor(s2, m, 32);
        }
        float mean = s * inv;
        float var  = s2 * inv - mean * mean;
        float rstd = rsqrtf(var + LN_EPS);
#pragma unroll
        for (int nt = 0; nt < 8; ++nt) {
            float x = (acc[nt][v] - mean) * rstd * gg[nt] + be[nt];
            acc[nt][v] = fmaxf(x, 0.0f);
        }
    }
}

// store 16x128 fp32 tile (WMMA layout) into LDS as bf16, row stride Kp halves
__device__ __forceinline__ void store_tile_bf16(__bf16* As, int Kp, int m0, int lane,
                                                v8f acc[8]) {
    const int col = lane & 15;
    const int lh  = lane >> 4;
#pragma unroll
    for (int v = 0; v < 8; ++v) {
        int row = m0 + v + 8 * lh;
#pragma unroll
        for (int nt = 0; nt < 8; ++nt)
            As[row * Kp + nt * 16 + col] = f2bf(acc[nt][v]);
    }
}

// ---------------- weight pre-pack: fp32 (K x 128) -> bf16 fragment order ----
__global__ void pack_w_kernel(const float* __restrict__ src, int Ktot, int kchunks,
                              __bf16* __restrict__ dst) {
    int p = blockIdx.x * 256 + threadIdx.x;
    int total = kchunks * 8 * 512;
    if (p >= total) return;
    int kc  = p >> 12;           // / (8*512)
    int rem = p & 4095;
    int nt  = rem >> 9;
    int q   = rem & 511;
    int lane = q >> 4;
    int hh   = q & 15;
    int col  = lane & 15;
    int lhi  = lane >> 4;
    int kr   = (hh & 7) + lhi * 8 + ((hh >> 3) << 4);
    int k    = kc * 32 + kr;
    int n    = nt * 16 + col;
    float v = (k < Ktot) ? src[(size_t)k * DIM + n] : 0.0f;
    dst[p] = f2bf(v);
}

// ---------------- init: h = emb[atoms], pos working copy --------------------
__global__ void init_kernel(const int* __restrict__ atoms, const float* __restrict__ emb,
                            const float* __restrict__ pos_in,
                            float* __restrict__ h, float* __restrict__ pos_cur) {
    int idx = blockIdx.x * 256 + threadIdx.x;
    if (idx < N_NODES * DIM) {
        int i = idx >> 7, d = idx & 127;
        h[idx] = emb[(size_t)atoms[i] * DIM + d];
    }
    if (idx < N_NODES * 3) pos_cur[idx] = pos_in[idx];
}

__global__ void zero_kernel(float* __restrict__ p, int n) {
    int i = blockIdx.x * 256 + threadIdx.x;
    if (i < n) p[i] = 0.0f;
}

// ---------------- fused edge kernel (message + pos MLPs + scatter) ----------
// 128 edges per block, 256 threads (8 wave32s); wave w owns rows w*16..w*16+15.
__global__ __launch_bounds__(256) void edge_kernel(
    const float* __restrict__ h, const float* __restrict__ pos,
    const int* __restrict__ src, const int* __restrict__ dst,
    const __bf16* __restrict__ w1p, const float* __restrict__ b1,
    const float* __restrict__ g1,  const float* __restrict__ B1,
    const __bf16* __restrict__ w2p, const float* __restrict__ b2,
    const float* __restrict__ g2,  const float* __restrict__ B2,
    const __bf16* __restrict__ q1p, const float* __restrict__ qb1,
    const float* __restrict__ qg1, const float* __restrict__ qB1,
    const float* __restrict__ qw2, const float* __restrict__ qb2,
    float* __restrict__ msg_aggr, float* __restrict__ pos_aggr,
    float* __restrict__ cnt)
{
    __shared__ __align__(16) __bf16 As[128 * KP1];   // 75776 B, reused per stage
    __shared__ float pdls[128 * 3];
    __shared__ float sbuf[128];
    __shared__ int   dsts[128];

    const int tid = threadIdx.x;
    const int e0  = blockIdx.x * 128;

    // ---- stage A1 = [h_dst | h_src | dist | 0-pad] as bf16 (stride KP1) ----
    {
        int r = tid >> 1, half = tid & 1;
        int e = e0 + r;
        bool valid = e < N_EDGES;
        int is = valid ? src[e] : 0;
        int id = valid ? dst[e] : 0;
        const float4* hp4 = (const float4*)(h + (size_t)(half ? is : id) * DIM);
        unsigned* arow32 = (unsigned*)(As + r * KP1 + half * DIM);
#pragma unroll 4
        for (int c = 0; c < DIM / 4; ++c) {
            float4 v4 = valid ? hp4[c] : make_float4(0.f, 0.f, 0.f, 0.f);
            arow32[2 * c]     = pack2bf(v4.x, v4.y);
            arow32[2 * c + 1] = pack2bf(v4.z, v4.w);
        }
        if (half) {
            float dx = 0.f, dy = 0.f, dz = 0.f, dist = 0.f;
            if (valid) {
                dx = pos[(size_t)id * 3 + 0] - pos[(size_t)is * 3 + 0];
                dy = pos[(size_t)id * 3 + 1] - pos[(size_t)is * 3 + 1];
                dz = pos[(size_t)id * 3 + 2] - pos[(size_t)is * 3 + 2];
                dist = sqrtf(dx * dx + dy * dy + dz * dz);
            }
            As[r * KP1 + 256] = f2bf(dist);
            // zero K = 257..287 (GEMM reads K < 288)
            unsigned* zp = (unsigned*)(As + r * KP1 + 258);
            As[r * KP1 + 257] = f2bf(0.0f);
#pragma unroll
            for (int c = 0; c < 15; ++c) zp[c] = 0u;
            pdls[r * 3 + 0] = dx; pdls[r * 3 + 1] = dy; pdls[r * 3 + 2] = dz;
            dsts[r] = valid ? id : -1;
        }
    }
    __syncthreads();

    const int lane = tid & 31;
    const int wave = tid >> 5;
    const int m0   = wave * 16;
    const int col  = lane & 15;
    const int lh   = lane >> 4;

    // ---- GEMM1: (128 x 257) @ mw1 -> LN -> ReLU ----
    v8f acc[8];
#pragma unroll
    for (int i = 0; i < 8; ++i)
#pragma unroll
        for (int v = 0; v < 8; ++v) acc[i][v] = 0.0f;
    wave_gemm(As, KP1, 9, w1p, m0, lane, acc);
    bias_ln_relu(acc, b1, g1, B1, lane);
    __syncthreads();
    store_tile_bf16(As, KP2, m0, lane, acc);
    __syncthreads();

    // ---- GEMM2: @ mw2 -> LN -> ReLU  => message m ----
#pragma unroll
    for (int i = 0; i < 8; ++i)
#pragma unroll
        for (int v = 0; v < 8; ++v) acc[i][v] = 0.0f;
    wave_gemm(As, KP2, 4, w2p, m0, lane, acc);
    bias_ln_relu(acc, b2, g2, B2, lane);
    __syncthreads();
    store_tile_bf16(As, KP2, m0, lane, acc);   // A3 input for pos MLP

    // scatter-sum m into msg_aggr[dst]
#pragma unroll
    for (int v = 0; v < 8; ++v) {
        int r  = m0 + v + 8 * lh;
        int id = dsts[r];
        if (id >= 0) {
            float* mp = msg_aggr + (size_t)id * DIM;
#pragma unroll
            for (int nt = 0; nt < 8; ++nt)
                atomicAdd(&mp[nt * 16 + col], acc[nt][v]);
        }
    }
    __syncthreads();

    // ---- GEMM3: pos MLP layer1 -> LN -> ReLU ----
    v8f acc3[8];
#pragma unroll
    for (int i = 0; i < 8; ++i)
#pragma unroll
        for (int v = 0; v < 8; ++v) acc3[i][v] = 0.0f;
    wave_gemm(As, KP2, 4, q1p, m0, lane, acc3);
    bias_ln_relu(acc3, qb1, qg1, qB1, lane);

    // s = t @ pw2 + pb2   (row dot over 128 cols)
    {
        float w2l[8];
#pragma unroll
        for (int nt = 0; nt < 8; ++nt) w2l[nt] = qw2[nt * 16 + col];
#pragma unroll
        for (int v = 0; v < 8; ++v) {
            float s = 0.f;
#pragma unroll
            for (int nt = 0; nt < 8; ++nt) s += acc3[nt][v] * w2l[nt];
            for (int m = 1; m < 16; m <<= 1) s += __shfl_xor(s, m, 32);
            if (col == 0) sbuf[m0 + v + 8 * lh] = s + qb2[0];
        }
    }
    __syncthreads();

    // scatter pd*s and counts
    if (tid < 128) {
        int id = dsts[tid];
        if (id >= 0) {
            float s = sbuf[tid];
            atomicAdd(&pos_aggr[(size_t)id * 3 + 0], pdls[tid * 3 + 0] * s);
            atomicAdd(&pos_aggr[(size_t)id * 3 + 1], pdls[tid * 3 + 1] * s);
            atomicAdd(&pos_aggr[(size_t)id * 3 + 2], pdls[tid * 3 + 2] * s);
            atomicAdd(&cnt[id], 1.0f);
        }
    }
}

// ---------------- fused node update kernel ---------------------------------
__global__ __launch_bounds__(256) void node_kernel(
    float* __restrict__ h, float* __restrict__ pos,
    const float* __restrict__ msg_aggr, const float* __restrict__ pos_aggr,
    const float* __restrict__ cnt,
    const __bf16* __restrict__ w1p, const float* __restrict__ b1,
    const float* __restrict__ g1,  const float* __restrict__ B1,
    const __bf16* __restrict__ w2p, const float* __restrict__ b2,
    const float* __restrict__ g2,  const float* __restrict__ B2)
{
    __shared__ __align__(16) __bf16 As[128 * KPN];   // 67584 B

    const int tid = threadIdx.x;
    const int n0  = blockIdx.x * 128;

    // stage A1 = [h | msg_aggr] (stride KPN)
    {
        int r = tid >> 1, half = tid & 1;
        int i = n0 + r;
        bool valid = i < N_NODES;
        int ii = valid ? i : 0;
        const float4* p4 = (const float4*)(half ? (msg_aggr + (size_t)ii * DIM)
                                                : (h + (size_t)ii * DIM));
        unsigned* arow32 = (unsigned*)(As + r * KPN + half * DIM);
#pragma unroll 4
        for (int c = 0; c < DIM / 4; ++c) {
            float4 v4 = valid ? p4[c] : make_float4(0.f, 0.f, 0.f, 0.f);
            arow32[2 * c]     = pack2bf(v4.x, v4.y);
            arow32[2 * c + 1] = pack2bf(v4.z, v4.w);
        }
    }
    __syncthreads();

    const int lane = tid & 31;
    const int wave = tid >> 5;
    const int m0   = wave * 16;
    const int col  = lane & 15;
    const int lh   = lane >> 4;

    v8f acc[8];
#pragma unroll
    for (int i = 0; i < 8; ++i)
#pragma unroll
        for (int v = 0; v < 8; ++v) acc[i][v] = 0.0f;
    wave_gemm(As, KPN, 8, w1p, m0, lane, acc);
    bias_ln_relu(acc, b1, g1, B1, lane);
    __syncthreads();
    store_tile_bf16(As, KP2, m0, lane, acc);
    __syncthreads();

#pragma unroll
    for (int i = 0; i < 8; ++i)
#pragma unroll
        for (int v = 0; v < 8; ++v) acc[i][v] = 0.0f;
    wave_gemm(As, KP2, 4, w2p, m0, lane, acc);
    bias_ln_relu(acc, b2, g2, B2, lane);

    // residual: h += u
#pragma unroll
    for (int v = 0; v < 8; ++v) {
        int r = m0 + v + 8 * lh;
        int i = n0 + r;
        if (i < N_NODES) {
            float* hp = h + (size_t)i * DIM;
#pragma unroll
            for (int nt = 0; nt < 8; ++nt)
                hp[nt * 16 + col] += acc[nt][v];
        }
    }

    // pos += pos_aggr / max(cnt, 1)
    if (tid < 128) {
        int i = n0 + tid;
        if (i < N_NODES) {
            float c = cnt[i]; c = fmaxf(c, 1.0f);
            float rinv = 1.0f / c;
            pos[(size_t)i * 3 + 0] += pos_aggr[(size_t)i * 3 + 0] * rinv;
            pos[(size_t)i * 3 + 1] += pos_aggr[(size_t)i * 3 + 1] * rinv;
            pos[(size_t)i * 3 + 2] += pos_aggr[(size_t)i * 3 + 2] * rinv;
        }
    }
}

// ---------------- pooling + head -------------------------------------------
__global__ void pool_kernel(const float* __restrict__ h, const int* __restrict__ batch,
                            float* __restrict__ pooled) {
    int idx = blockIdx.x * 256 + threadIdx.x;
    if (idx < N_NODES * DIM) {
        int i = idx >> 7, d = idx & 127;
        atomicAdd(&pooled[(size_t)batch[i] * DIM + d], h[idx]);
    }
}

__global__ __launch_bounds__(64) void head_kernel(
    const float* __restrict__ pooled,
    const float* __restrict__ rw1, const float* __restrict__ rb1,
    const float* __restrict__ rw2, const float* __restrict__ rb2,
    float* __restrict__ out) {
    int g = threadIdx.x;            // 0..63
    float accum = 0.0f;
    for (int n = 0; n < DIM; ++n) {
        float t = rb1[n];
        for (int k = 0; k < DIM; ++k)
            t += pooled[(size_t)g * DIM + k] * rw1[(size_t)k * DIM + n];
        t = fmaxf(t, 0.0f);
        accum += t * rw2[n];
    }
    out[g] = accum + rb2[0];
}

// ---------------------------------------------------------------------------
extern "C" void kernel_launch(void* const* d_in, const int* in_sizes, int n_in,
                              void* d_out, int out_size, void* d_ws, size_t ws_size,
                              hipStream_t stream) {
    const int*   atoms = (const int*)  d_in[0];
    const float* pos   = (const float*)d_in[1];
    const int*   eidx  = (const int*)  d_in[2];
    const int*   batch = (const int*)  d_in[3];
    const float* emb   = (const float*)d_in[4];
    const float* mw1 = (const float*)d_in[5],  *mb1 = (const float*)d_in[6];
    const float* mg1 = (const float*)d_in[7],  *mB1 = (const float*)d_in[8];
    const float* mw2 = (const float*)d_in[9],  *mb2 = (const float*)d_in[10];
    const float* mg2 = (const float*)d_in[11], *mB2 = (const float*)d_in[12];
    const float* pw1 = (const float*)d_in[13], *pb1 = (const float*)d_in[14];
    const float* pg1 = (const float*)d_in[15], *pB1 = (const float*)d_in[16];
    const float* pw2 = (const float*)d_in[17], *pb2 = (const float*)d_in[18];
    const float* uw1 = (const float*)d_in[19], *ub1 = (const float*)d_in[20];
    const float* ug1 = (const float*)d_in[21], *uB1 = (const float*)d_in[22];
    const float* uw2 = (const float*)d_in[23], *ub2 = (const float*)d_in[24];
    const float* ug2 = (const float*)d_in[25], *uB2 = (const float*)d_in[26];
    const float* rw1 = (const float*)d_in[27], *rb1 = (const float*)d_in[28];
    const float* rw2 = (const float*)d_in[29], *rb2 = (const float*)d_in[30];
    float* out = (float*)d_out;

    const int* esrc = eidx;              // edge_index[0]
    const int* edst = eidx + N_EDGES;    // edge_index[1]

    // ---- workspace carve-up ----
    char* ws = (char*)d_ws;
    size_t off = 0;
    auto ALLOC = [&](size_t bytes) -> void* {
        void* p = ws + off;
        off += (bytes + 255) & ~(size_t)255;
        return p;
    };
    float* hbuf   = (float*)ALLOC((size_t)N_NODES * DIM * sizeof(float));
    float* msg    = (float*)ALLOC((size_t)N_NODES * DIM * sizeof(float));
    float* posc   = (float*)ALLOC((size_t)N_NODES * 3 * sizeof(float));
    float* posa   = (float*)ALLOC((size_t)N_NODES * 3 * sizeof(float));
    float* cnt    = (float*)ALLOC((size_t)N_NODES * sizeof(float));
    float* pooled = (float*)ALLOC((size_t)NGRAPH * DIM * sizeof(float));

    const size_t L_M1 = 9 * 8 * 512;  // mw1 packed elems (K padded 257->288)
    const size_t L_M2 = 4 * 8 * 512;
    const size_t L_P1 = 4 * 8 * 512;
    const size_t L_U1 = 8 * 8 * 512;
    const size_t L_U2 = 4 * 8 * 512;
    const size_t LTOT = L_M1 + L_M2 + L_P1 + L_U1 + L_U2;
    __bf16* wpack = (__bf16*)ALLOC(LTOT * NLAYERS * sizeof(__bf16));

    // ---- pre-pack weights to bf16 fragment layout ----
    for (int l = 0; l < NLAYERS; ++l) {
        __bf16* base = wpack + (size_t)l * LTOT;
        pack_w_kernel<<<(int)((L_M1 + 255) / 256), 256, 0, stream>>>(
            mw1 + (size_t)l * 257 * DIM, 257, 9, base);
        pack_w_kernel<<<(int)((L_M2 + 255) / 256), 256, 0, stream>>>(
            mw2 + (size_t)l * DIM * DIM, 128, 4, base + L_M1);
        pack_w_kernel<<<(int)((L_P1 + 255) / 256), 256, 0, stream>>>(
            pw1 + (size_t)l * DIM * DIM, 128, 4, base + L_M1 + L_M2);
        pack_w_kernel<<<(int)((L_U1 + 255) / 256), 256, 0, stream>>>(
            uw1 + (size_t)l * 256 * DIM, 256, 8, base + L_M1 + L_M2 + L_P1);
        pack_w_kernel<<<(int)((L_U2 + 255) / 256), 256, 0, stream>>>(
            uw2 + (size_t)l * DIM * DIM, 128, 4, base + L_M1 + L_M2 + L_P1 + L_U1);
    }

    // ---- h = emb[atoms]; pos working copy ----
    init_kernel<<<(N_NODES * DIM + 255) / 256, 256, 0, stream>>>(atoms, emb, pos, hbuf, posc);

    // ---- layers ----
    for (int l = 0; l < NLAYERS; ++l) {
        zero_kernel<<<(N_NODES * DIM + 255) / 256, 256, 0, stream>>>(msg, N_NODES * DIM);
        zero_kernel<<<(N_NODES * 3 + 255) / 256, 256, 0, stream>>>(posa, N_NODES * 3);
        zero_kernel<<<(N_NODES + 255) / 256, 256, 0, stream>>>(cnt, N_NODES);

        __bf16* base = wpack + (size_t)l * LTOT;
        edge_kernel<<<(N_EDGES + 127) / 128, 256, 0, stream>>>(
            hbuf, posc, esrc, edst,
            base,                    mb1 + (size_t)l * DIM, mg1 + (size_t)l * DIM, mB1 + (size_t)l * DIM,
            base + L_M1,             mb2 + (size_t)l * DIM, mg2 + (size_t)l * DIM, mB2 + (size_t)l * DIM,
            base + L_M1 + L_M2,      pb1 + (size_t)l * DIM, pg1 + (size_t)l * DIM, pB1 + (size_t)l * DIM,
            pw2 + (size_t)l * DIM,   pb2 + l,
            msg, posa, cnt);

        node_kernel<<<(N_NODES + 127) / 128, 256, 0, stream>>>(
            hbuf, posc, msg, posa, cnt,
            base + L_M1 + L_M2 + L_P1,        ub1 + (size_t)l * DIM, ug1 + (size_t)l * DIM, uB1 + (size_t)l * DIM,
            base + L_M1 + L_M2 + L_P1 + L_U1, ub2 + (size_t)l * DIM, ug2 + (size_t)l * DIM, uB2 + (size_t)l * DIM);
    }

    // ---- pool + head ----
    zero_kernel<<<(NGRAPH * DIM + 255) / 256, 256, 0, stream>>>(pooled, NGRAPH * DIM);
    pool_kernel<<<(N_NODES * DIM + 255) / 256, 256, 0, stream>>>(hbuf, batch, pooled);
    head_kernel<<<1, 64, 0, stream>>>(pooled, rw1, rb1, rw2, rb2, out);
}
